// CausalSelfAttention_59554016526890
// MI455X (gfx1250) — compile-verified, matrix operand order checked by reading, not run on hardware
//
#include <hip/hip_runtime.h>

// ---------------------------------------------------------------------------
// CDNA5 (gfx1250) causal self-attention, bf16 WMMA path, double-buffered LDS,
// async global->LDS staging (GLOBAL_LOAD_ASYNC_TO_LDS_B128 / ASYNCcnt).
// B=4, N=2048, E=2048, D=2048, 3D=6144.
// ws layout: Qbf | Kbf | Vbf (bf16, BND each) | S (f32, BNN) | P (bf16, BNN)
// d_out: y (f32, B*N*D) then denom (f32, B*N)
//
// GEMM scheme: 256 threads = 8 wave32, block tile 128(M) x 256(N), kTile=32
// (== WMMA K). Waves arranged 2(M) x 4(N); each wave owns a 64x64 tile =
// 4x4 WMMA subtiles -> 16 v_wmma per k-step against 16 ds_load_b128.
// LDS ping-pong buffers overlap next-tile global loads with current WMMAs.
// ---------------------------------------------------------------------------

#if __has_builtin(__builtin_amdgcn_global_load_async_to_lds_b128) && \
    __has_builtin(__builtin_amdgcn_s_wait_asynccnt)
#define HAVE_ASYNC_LDS 1
#else
#define HAVE_ASYNC_LDS 0
#endif

typedef __attribute__((ext_vector_type(16))) __bf16        v16bf;
typedef __attribute__((ext_vector_type(8)))  float         v8f;
typedef __attribute__((ext_vector_type(4)))  float         v4f;
typedef __attribute__((ext_vector_type(4)))  unsigned int  v4u;
typedef __attribute__((ext_vector_type(2)))  unsigned int  v2u;
typedef __attribute__((ext_vector_type(4)))  int           v4i;

#define AS_GLOBAL __attribute__((address_space(1)))
#define AS_LDS    __attribute__((address_space(3)))

#define Bq   4
#define Nq   2048
#define Eq   2048
#define Dq   2048
#define TDq  6144
#define LDSP 40        // LDS row pitch in bf16 elems (32 + 8 pad)

__device__ inline unsigned short f32_to_bf16(float f) {
  union { float f; unsigned int u; } c; c.f = f;
  unsigned int u = c.u;
  u += 0x7FFFu + ((u >> 16) & 1u);          // round-to-nearest-even
  return (unsigned short)(u >> 16);
}

union Frag { v16bf v; v4u q[2]; };

// Fragment from LDS tile, row-major [row][LDSP], K contiguous.
// ISA 16-bit A 16x32 layout: lanes 0-15: K=0..7 (VGPR0-3), K=16..23 (VGPR4-7);
// lanes 16-31: K=8..15, K=24..31. B mirrors with lane = column.
__device__ inline Frag load_frag(const unsigned short* base, int row, int half) {
  Frag f;
  const unsigned short* p = base + row * LDSP + half * 8;
  f.q[0] = *(const v4u*)(p);
  f.q[1] = *(const v4u*)(p + 16);
  return f;
}

__device__ inline void load_frags(const unsigned short* As, const unsigned short* Bs,
                                  int wm, int wn, int lmod, int half,
                                  Frag a[4], Frag b[4]) {
#pragma unroll
  for (int i = 0; i < 4; ++i) {
    a[i] = load_frag(As, wm * 64 + i * 16 + lmod, half);
    b[i] = load_frag(Bs, wn * 64 + i * 16 + lmod, half);
  }
}

__device__ inline void wmma16(const Frag a[4], const Frag b[4], v8f acc[4][4]) {
#pragma unroll
  for (int mi = 0; mi < 4; ++mi)
#pragma unroll
    for (int ni = 0; ni < 4; ++ni)
      acc[mi][ni] = __builtin_amdgcn_wmma_f32_16x16x32_bf16(
          false, a[mi].v, false, b[ni].v, (short)0, acc[mi][ni], false, false);
}

// ---- LDS staging helpers (256 threads) ------------------------------------

// Pure-copy staging, bf16 row-major: dst[r][c] = src[(row0+r)*ld + col0+c].
// Uses GLOBAL_LOAD_ASYNC_TO_LDS_B128 (ASYNCcnt) when the toolchain has it.
template <int ROWS>
__device__ inline void stage_bf16(unsigned short* dst, const unsigned short* src,
                                  size_t ld, size_t row0, size_t col0, int tid) {
#pragma unroll
  for (int p = 0; p < ROWS / 64; ++p) {
    int r = (tid >> 2) + p * 64;
    int c = (tid & 3) * 8;
#if HAVE_ASYNC_LDS
    __builtin_amdgcn_global_load_async_to_lds_b128(
        (AS_GLOBAL v4i*)(src + (row0 + r) * ld + col0 + c),
        (AS_LDS v4i*)(dst + r * LDSP + c),
        0, 0);
#else
    *(v4u*)(dst + r * LDSP + c) = *(const v4u*)(src + (row0 + r) * ld + col0 + c);
#endif
  }
}

// Fence async staging before the workgroup barrier (compiler barriers only
// track LOADcnt/DScnt, not ASYNCcnt).
__device__ inline void stage_fence() {
#if HAVE_ASYNC_LDS
  __builtin_amdgcn_s_wait_asynccnt(0);
#endif
}

// bf16 transpose, 256 n x 32 k: dst[n][k] = src[(k0+k)*ld + n0+n].
// Loads two adjacent k-rows and packs (k,k+1) pairs -> dword LDS stores.
__device__ inline void stage_bf16_T256(unsigned short* dst, const unsigned short* src,
                                       size_t ld, size_t k0, size_t n0, int tid) {
#pragma unroll
  for (int p = 0; p < 2; ++p) {
    int k  = ((tid >> 5) + p * 8) * 2;   // 0,2,...,30
    int nn = (tid & 31) * 8;             // 0..248
    v4u d0 = *(const v4u*)(src + (k0 + k) * ld + n0 + nn);
    v4u d1 = *(const v4u*)(src + (k0 + k + 1) * ld + n0 + nn);
    const unsigned short* e0 = (const unsigned short*)&d0;
    const unsigned short* e1 = (const unsigned short*)&d1;
#pragma unroll
    for (int i = 0; i < 8; ++i) {
      unsigned int w = (unsigned)e0[i] | ((unsigned)e1[i] << 16);
      *(unsigned int*)(dst + (nn + i) * LDSP + k) = w;
    }
  }
}

// f32 -> bf16 row-major, 128 x 32 tile
__device__ inline void stage_f32(unsigned short* dst, const float* src,
                                 size_t ld, size_t row0, size_t col0, int tid) {
#pragma unroll
  for (int p = 0; p < 4; ++p) {
    int r = (tid >> 3) + p * 32;
    int c = (tid & 7) * 4;
    v4f d = *(const v4f*)(src + (row0 + r) * ld + col0 + c);
    v2u pk;
    pk[0] = (unsigned)f32_to_bf16(d[0]) | ((unsigned)f32_to_bf16(d[1]) << 16);
    pk[1] = (unsigned)f32_to_bf16(d[2]) | ((unsigned)f32_to_bf16(d[3]) << 16);
    *(v2u*)(dst + r * LDSP + c) = pk;
  }
}

// f32 -> bf16 transpose, 256 n x 32 k: dst[n][k] = src[(k0+k)*ld + n0+n]
__device__ inline void stage_f32_T256(unsigned short* dst, const float* src,
                                      size_t ld, size_t k0, size_t n0, int tid) {
#pragma unroll
  for (int p = 0; p < 4; ++p) {
    int k  = ((tid >> 6) + p * 4) * 2;   // 0,2,...,30
    int nn = (tid & 63) * 4;             // 0..252
    v4f d0 = *(const v4f*)(src + (k0 + k) * ld + n0 + nn);
    v4f d1 = *(const v4f*)(src + (k0 + k + 1) * ld + n0 + nn);
#pragma unroll
    for (int i = 0; i < 4; ++i) {
      unsigned int w = (unsigned)f32_to_bf16(d0[i]) |
                       ((unsigned)f32_to_bf16(d1[i]) << 16);
      *(unsigned int*)(dst + (nn + i) * LDSP + k) = w;
    }
  }
}

#define GEMM_PREAMBLE()                                            \
  const int tid  = threadIdx.x;                                    \
  const int lane = tid & 31, wave = tid >> 5;                      \
  const int half = lane >> 4, lmod = lane & 15;                    \
  const int wm = wave & 1, wn = wave >> 1;                         \
  v8f acc[4][4];                                                   \
  {                                                                \
    const v8f vz = {0.f,0.f,0.f,0.f,0.f,0.f,0.f,0.f};              \
    _Pragma("unroll") for (int mi = 0; mi < 4; ++mi)               \
      _Pragma("unroll") for (int ni = 0; ni < 4; ++ni)             \
        acc[mi][ni] = vz;                                          \
  }

// ---------------------------------------------------------------------------
// Kernel 1: qkv = x @ W + b  -> split into bf16 Q|K|V  ([B*N] x [3D], K=E)
// grid (TD/256, B*N/128)
// ---------------------------------------------------------------------------
__global__ void __launch_bounds__(256)
qkv_gemm_kernel(const float* __restrict__ x, const float* __restrict__ W,
                const float* __restrict__ bias, unsigned short* __restrict__ QKV) {
  __shared__ __align__(16) unsigned short As[2][128 * LDSP];
  __shared__ __align__(16) unsigned short Bs[2][256 * LDSP];
  GEMM_PREAMBLE();
  const size_t row0 = (size_t)blockIdx.y * 128;   // flat b*N + n
  const size_t col0 = (size_t)blockIdx.x * 256;   // 0..6143

  stage_f32(As[0], x, Eq, row0, 0, tid);
  stage_f32_T256(Bs[0], W, TDq, 0, col0, tid);
  __syncthreads();
  int cur = 0;
  for (int kk = 32; kk < Eq; kk += 32) {
    Frag a[4], b[4];
    load_frags(As[cur], Bs[cur], wm, wn, lmod, half, a, b);
    if (kk + 32 < Eq)
      __builtin_prefetch(x + (row0 + (size_t)(tid >> 1)) * Eq + kk + 32 + (tid & 1) * 16, 0, 3);
    stage_f32(As[cur ^ 1], x, Eq, row0, (size_t)kk, tid);
    stage_f32_T256(Bs[cur ^ 1], W, TDq, (size_t)kk, col0, tid);
    wmma16(a, b, acc);
    __syncthreads();
    cur ^= 1;
  }
  {
    Frag a[4], b[4];
    load_frags(As[cur], Bs[cur], wm, wn, lmod, half, a, b);
    wmma16(a, b, acc);
  }

  const size_t BND = (size_t)Bq * Nq * Dq;
#pragma unroll
  for (int mi = 0; mi < 4; ++mi)
#pragma unroll
    for (int ni = 0; ni < 4; ++ni) {
      const size_t col = col0 + wn * 64 + ni * 16 + lmod;
      const float bc = bias[col];
      const size_t which = col >> 11;           // 0=Q 1=K 2=V
      const size_t d = col & 2047;
      unsigned short* dst = QKV + which * BND + d;
#pragma unroll
      for (int r = 0; r < 8; ++r) {
        const size_t row = row0 + wm * 64 + mi * 16 + 8 * half + r;
        dst[row * Dq] = f32_to_bf16(acc[mi][ni][r] + bc);
      }
    }
}

// ---------------------------------------------------------------------------
// Kernel 2: S = scale * Q @ K^T, causal mask, lower-triangular tiles only.
// grid (N/256 keys, N/128 queries, B)
// ---------------------------------------------------------------------------
__global__ void __launch_bounds__(256)
attn_scores_kernel(const unsigned short* __restrict__ Qb,
                   const unsigned short* __restrict__ Kb,
                   float* __restrict__ S, float scale) {
  const int kt = blockIdx.x, qt = blockIdx.y, bz = blockIdx.z;
  if (2 * kt > qt) return;                   // block-uniform: EXEC stays all-1
  __shared__ __align__(16) unsigned short As[2][128 * LDSP];
  __shared__ __align__(16) unsigned short Bs[2][256 * LDSP];
  GEMM_PREAMBLE();
  const size_t q0 = (size_t)qt * 128, k0 = (size_t)kt * 256;
  const unsigned short* Q = Qb + (size_t)bz * Nq * Dq;
  const unsigned short* K = Kb + (size_t)bz * Nq * Dq;

  stage_bf16<128>(As[0], Q, Dq, q0, 0, tid);
  stage_bf16<256>(Bs[0], K, Dq, k0, 0, tid);   // n-major B == K row-major
  stage_fence();
  __syncthreads();
  int cur = 0;
  for (int kk = 32; kk < Dq; kk += 32) {
    Frag a[4], b[4];
    load_frags(As[cur], Bs[cur], wm, wn, lmod, half, a, b);
    if (kk + 32 < Dq)
      __builtin_prefetch(Q + (q0 + (size_t)(tid >> 1)) * Dq + kk + 32 + (tid & 1) * 16, 0, 3);
    stage_bf16<128>(As[cur ^ 1], Q, Dq, q0, (size_t)kk, tid);
    stage_bf16<256>(Bs[cur ^ 1], K, Dq, k0, (size_t)kk, tid);
    wmma16(a, b, acc);
    stage_fence();
    __syncthreads();
    cur ^= 1;
  }
  {
    Frag a[4], b[4];
    load_frags(As[cur], Bs[cur], wm, wn, lmod, half, a, b);
    wmma16(a, b, acc);
  }

  float* Srow = S + (size_t)bz * Nq * Nq;
#pragma unroll
  for (int mi = 0; mi < 4; ++mi)
#pragma unroll
    for (int ni = 0; ni < 4; ++ni) {
      const size_t kcol = k0 + wn * 64 + ni * 16 + lmod;
#pragma unroll
      for (int r = 0; r < 8; ++r) {
        const size_t qrow = q0 + wm * 64 + mi * 16 + 8 * half + r;
        float s = acc[mi][ni][r] * scale;
        if (kcol > qrow) s = -__builtin_inff();
        Srow[qrow * Nq + kcol] = s;
      }
    }
}

// ---------------------------------------------------------------------------
// Kernel 3: per-row stable softmax -> P (bf16, zero-filled to 128-key tile),
//           denom (f32) to output tail.  grid (N, B), 256 threads
// ---------------------------------------------------------------------------
__global__ void __launch_bounds__(256)
softmax_denom_kernel(const float* __restrict__ S, unsigned short* __restrict__ P,
                     float* __restrict__ denomOut) {
  const int q = blockIdx.x, bz = blockIdx.y, tid = threadIdx.x;
  const float* srow = S + ((size_t)bz * Nq + q) * Nq;
  unsigned short* prow = P + ((size_t)bz * Nq + q) * Nq;
  const int L = q + 1;
  const int kend = ((q >> 7) + 1) << 7;      // pad to 128-tile for kernel 4
  __shared__ float red[256];

  float m = -__builtin_inff();
  for (int k = tid; k < L; k += 256) m = fmaxf(m, srow[k]);
  red[tid] = m; __syncthreads();
  for (int s = 128; s > 0; s >>= 1) {
    if (tid < s) red[tid] = fmaxf(red[tid], red[tid + s]);
    __syncthreads();
  }
  m = red[0]; __syncthreads();

  float sum = 0.f;
  for (int k = tid; k < L; k += 256) sum += __expf(srow[k] - m);
  red[tid] = sum; __syncthreads();
  for (int s = 128; s > 0; s >>= 1) {
    if (tid < s) red[tid] += red[tid + s];
    __syncthreads();
  }
  const float denom = red[0];
  const float inv = 1.0f / denom;

  for (int k = tid; k < kend; k += 256)
    prow[k] = (k < L) ? f32_to_bf16(__expf(srow[k] - m) * inv) : (unsigned short)0;
  if (tid == 0) denomOut[(size_t)bz * Nq + q] = denom;
}

// ---------------------------------------------------------------------------
// Kernel 4: y = P @ V  (causal-truncated K loop).  grid (D/256, N/128, B)
// ---------------------------------------------------------------------------
__global__ void __launch_bounds__(256)
attn_pv_kernel(const unsigned short* __restrict__ Pb,
               const unsigned short* __restrict__ Vb,
               float* __restrict__ Y) {
  const int dt = blockIdx.x, qt = blockIdx.y, bz = blockIdx.z;
  __shared__ __align__(16) unsigned short As[2][128 * LDSP];
  __shared__ __align__(16) unsigned short Bs[2][256 * LDSP];
  GEMM_PREAMBLE();
  const size_t q0 = (size_t)qt * 128, d0 = (size_t)dt * 256;
  const unsigned short* Pm = Pb + (size_t)bz * Nq * Nq;
  const unsigned short* V  = Vb + (size_t)bz * Nq * Dq;

  const int kmax = (qt + 1) * 128;           // causal: keys <= query tile end
  stage_bf16<128>(As[0], Pm, Nq, q0, 0, tid);
  stage_bf16_T256(Bs[0], V, Dq, 0, d0, tid);  // Bs[d][key] = V[key][d]
  stage_fence();
  __syncthreads();
  int cur = 0;
  for (int kk = 32; kk < kmax; kk += 32) {
    Frag a[4], b[4];
    load_frags(As[cur], Bs[cur], wm, wn, lmod, half, a, b);
    if (kk + 32 < kmax)
      __builtin_prefetch(Pm + (q0 + (size_t)(tid >> 1)) * Nq + kk + 32 + (tid & 1) * 16, 0, 3);
    stage_bf16<128>(As[cur ^ 1], Pm, Nq, q0, (size_t)kk, tid);
    stage_bf16_T256(Bs[cur ^ 1], V, Dq, (size_t)kk, d0, tid);
    wmma16(a, b, acc);
    stage_fence();
    __syncthreads();
    cur ^= 1;
  }
  {
    Frag a[4], b[4];
    load_frags(As[cur], Bs[cur], wm, wn, lmod, half, a, b);
    wmma16(a, b, acc);
  }

  float* Yb = Y + (size_t)bz * Nq * Dq;
#pragma unroll
  for (int mi = 0; mi < 4; ++mi)
#pragma unroll
    for (int ni = 0; ni < 4; ++ni) {
      const size_t dcol = d0 + wn * 64 + ni * 16 + lmod;
#pragma unroll
      for (int r = 0; r < 8; ++r) {
        const size_t qrow = q0 + wm * 64 + mi * 16 + 8 * half + r;
        Yb[qrow * Dq + dcol] = acc[mi][ni][r];
      }
    }
}

// ---------------------------------------------------------------------------
extern "C" void kernel_launch(void* const* d_in, const int* in_sizes, int n_in,
                              void* d_out, int out_size, void* d_ws, size_t ws_size,
                              hipStream_t stream) {
  const float* x    = (const float*)d_in[0];   // [B,N,E]
  const float* W    = (const float*)d_in[1];   // [E,3D]
  const float* bias = (const float*)d_in[2];   // [3D]
  float* out = (float*)d_out;

  const size_t BND = (size_t)Bq * Nq * Dq;     // 16,777,216
  char* ws = (char*)d_ws;
  unsigned short* Qb = (unsigned short*)ws;            // bf16
  unsigned short* Kb = Qb + BND;
  unsigned short* Vb = Kb + BND;
  float*          S  = (float*)(ws + 3 * BND * 2);     // f32 scores
  unsigned short* P  = (unsigned short*)(ws + 3 * BND * 2 + BND * 4);

  float* y     = out;          // B*N*D
  float* denom = out + BND;    // B*N

  const float scale = 0.022097086912079608f;   // 1/sqrt(2048)

  qkv_gemm_kernel<<<dim3(TDq / 256, (Bq * Nq) / 128), 256, 0, stream>>>(x, W, bias, Qb);
  attn_scores_kernel<<<dim3(Nq / 256, Nq / 128, Bq), 256, 0, stream>>>(Qb, Kb, S, scale);
  softmax_denom_kernel<<<dim3(Nq, Bq), 256, 0, stream>>>(S, P, denom);
  attn_pv_kernel<<<dim3(Dq / 256, Nq / 128, Bq), 256, 0, stream>>>(P, Vb, y);
}
